// SelfAttention_36850819400284
// MI455X (gfx1250) — compile-verified, matrix operand order checked by reading, not run on hardware
//
#include <hip/hip_runtime.h>

// ---------------------------------------------------------------------------
// Self-attention for MI455X (gfx1250, wave32, WMMA + async global->LDS).
//   B=4, S=2048, D=1024, H=16, Dh=64.  All matmuls: v_wmma_f32_16x16x32_bf16,
//   fp32 accumulation. Intermediates bf16 in ws (72 MiB, L2-resident):
//   Q,K [B,H,S,Dh], V^T [B,H,Dh,S], attn-out [B,S,D], plus bf16 weights.
// ---------------------------------------------------------------------------

typedef __attribute__((ext_vector_type(16))) __bf16 v16bf;
typedef __attribute__((ext_vector_type(8)))  __bf16 v8bf;
typedef __attribute__((ext_vector_type(8)))  float  v8f;

// Async-copy builtin operand types: v4i pointers in addrspace(1)/(3).
typedef int v4i_ __attribute__((vector_size(16)));
typedef __attribute__((address_space(1))) v4i_ as1_v4i;
typedef __attribute__((address_space(3))) v4i_ as3_v4i;

#define WMMA_BF16(a, b, c)                                                   \
  __builtin_amdgcn_wmma_f32_16x16x32_bf16(false, (a), false, (b), (short)0,  \
                                          (c), false, false)

// 16B-granule async copy global -> LDS (ASYNCcnt-tracked DMA on gfx1250).
// Generic->as(1) pointer value is identity; generic LDS pointer's low 32 bits
// are the DS offset (flat-aperture rule), so integer round-trips are safe.
static __device__ inline void async_cp16(__bf16* dst_lds, const __bf16* src) {
#if __has_builtin(__builtin_amdgcn_global_load_async_to_lds_b128)
  __builtin_amdgcn_global_load_async_to_lds_b128(
      (as1_v4i*)(unsigned long long)src,
      (as3_v4i*)(unsigned int)(unsigned long long)dst_lds, 0, 0);
#else
  typedef __attribute__((ext_vector_type(4))) unsigned int u128;
  *(u128*)dst_lds = *(const u128*)src;
#endif
}

static __device__ inline void async_wait_all() {
#if __has_builtin(__builtin_amdgcn_s_wait_asynccnt)
  __builtin_amdgcn_s_wait_asynccnt(0);
#else
  asm volatile("s_wait_asynccnt 0x0" ::: "memory");
#endif
}

// 16-bit A-fragment (16x32), per ISA: lane half hh holds K = hh*8+[0..7] and
// K = 16+hh*8+[0..7] as two contiguous 16-byte runs.  B mirrors A.
static __device__ inline v16bf frag_from_bf16(const __bf16* rowp, int hh) {
  v16bf r;
  const __bf16* p0 = rowp + hh * 8;
  const __bf16* p1 = rowp + 16 + hh * 8;
#pragma unroll
  for (int i = 0; i < 8; ++i) { r[i] = p0[i]; r[8 + i] = p1[i]; }
  return r;
}

static __device__ inline v16bf frag_from_f32(const float* rowp, int hh) {
  v16bf r;
  const float* p0 = rowp + hh * 8;
  const float* p1 = rowp + 16 + hh * 8;
#pragma unroll
  for (int i = 0; i < 8; ++i) {
    r[i]     = (__bf16)p0[i];
    r[8 + i] = (__bf16)p1[i];
  }
  return r;
}

// ---------------------------------------------------------------------------
// One-time fp32 -> bf16 conversion (weights): each thread converts 8 elems.
// ---------------------------------------------------------------------------
__global__ __launch_bounds__(256) void cvt_f32_bf16(
    const float* __restrict__ src, __bf16* __restrict__ dst, int n) {
  const int i = (blockIdx.x * 256 + threadIdx.x) * 8;
  if (i >= n) return;
  v8bf o;
#pragma unroll
  for (int j = 0; j < 8; ++j) o[j] = (__bf16)src[i + j];
  *(v8bf*)(dst + i) = o;
}

// ---------------------------------------------------------------------------
// GEMM: out = A @ W^T + bias.  A:[M,K] f32|bf16, W:[N,K] bf16 (pre-converted).
// OUT_MODE 0: f32 row-major [M,N]           (final projection -> d_out)
// OUT_MODE 1: bf16 split-heads [B,H,S,Dh]   (Q, K intermediates)
// OUT_MODE 2: bf16 transposed  [B,H,Dh,S]   (V intermediate, for PV matmul)
// Block = 256 threads = 8 waves; wave tile 64x32 (4x2 WMMA); block 128x128.
// ---------------------------------------------------------------------------
template <bool A_BF16, int OUT_MODE>
__global__ __launch_bounds__(256) void gemm_wmma(
    const void* __restrict__ Ap, const __bf16* __restrict__ W,
    const float* __restrict__ bias, void* __restrict__ outp,
    int M, int N, int K) {
  const int lane  = threadIdx.x & 31;
  const int wave  = threadIdx.x >> 5;
  const int lsub  = lane & 15;
  const int khalf = lane >> 4;

  const int wm = blockIdx.x * 128 + (wave & 1) * 64;
  const int wn = blockIdx.y * 128 + (wave >> 1) * 32;

  // C/D layout: lane's N = lsub (constant) -> bias broadcasts into all 8 rows.
  v8f acc[4][2];
  const float b0 = bias[wn + lsub];
  const float b1 = bias[wn + 16 + lsub];
#pragma unroll
  for (int mi = 0; mi < 4; ++mi)
#pragma unroll
    for (int r = 0; r < 8; ++r) { acc[mi][0][r] = b0; acc[mi][1][r] = b1; }

  const float*  Af = (const float*)Ap;
  const __bf16* Ab = (const __bf16*)Ap;

  for (int kk = 0; kk < K; kk += 32) {
    v16bf bf0 = frag_from_bf16(W + (size_t)(wn + lsub)      * K + kk, khalf);
    v16bf bf1 = frag_from_bf16(W + (size_t)(wn + 16 + lsub) * K + kk, khalf);
    v16bf a[4];
#pragma unroll
    for (int mi = 0; mi < 4; ++mi) {
      const size_t row = (size_t)(wm + mi * 16 + lsub);
      if (A_BF16) a[mi] = frag_from_bf16(Ab + row * K + kk, khalf);
      else        a[mi] = frag_from_f32 (Af + row * K + kk, khalf);
    }
#pragma unroll
    for (int mi = 0; mi < 4; ++mi) {
      acc[mi][0] = WMMA_BF16(a[mi], bf0, acc[mi][0]);
      acc[mi][1] = WMMA_BF16(a[mi], bf1, acc[mi][1]);
    }
  }

#pragma unroll
  for (int mi = 0; mi < 4; ++mi)
#pragma unroll
    for (int ni = 0; ni < 2; ++ni)
#pragma unroll
      for (int r = 0; r < 8; ++r) {
        const int m = wm + mi * 16 + r + 8 * khalf;
        const int n = wn + ni * 16 + lsub;
        const float v = acc[mi][ni][r];
        if (OUT_MODE == 0) {
          ((float*)outp)[(size_t)m * N + n] = v;
        } else {
          const int b  = m >> 11, s  = m & 2047;  // S = 2048
          const int h  = n >> 6,  dh = n & 63;    // Dh = 64
          __bf16* o = (__bf16*)outp;
          if (OUT_MODE == 1)
            o[(((size_t)(b * 16 + h)) * 2048 + s) * 64 + dh] = (__bf16)v;
          else
            o[(((size_t)(b * 16 + h)) * 64 + dh) * 2048 + s] = (__bf16)v;
        }
      }
}

// ---------------------------------------------------------------------------
// Flash attention. 8 waves/block share one (b,h); K/V 32-key chunks are
// staged once per block into double-buffered LDS via async global->LDS DMA
// (ASYNCcnt), overlapping the next chunk's copy with this chunk's compute.
// Per chunk: 4 WMMA for S = Q K^T, online softmax (xor-shuffle row reduce),
// P via per-wave LDS tile (C-layout -> A-frag), 4 WMMA for O += P V.
// ---------------------------------------------------------------------------
#define S_LEN 2048
#define DH    64

__global__ __launch_bounds__(256) void attn_wmma(
    const __bf16* __restrict__ qb, const __bf16* __restrict__ kb,
    const __bf16* __restrict__ vt, const int* __restrict__ mask,
    __bf16* __restrict__ xb) {
  __shared__ __bf16 kt[2][32 * DH];    // [key][dh]  4 KiB per buffer
  __shared__ __bf16 vs[2][DH * 32];    // [dh][key]  4 KiB per buffer
  __shared__ __bf16 plds[8][16 * 32];  // per-wave P tile, 1 KiB each

  const int tid   = threadIdx.x;
  const int lane  = tid & 31;
  const int wave  = tid >> 5;
  const int lsub  = lane & 15;
  const int khalf = lane >> 4;

  const int bh   = blockIdx.x >> 4;                  // 0..63  (b*16 + h)
  const int qblk = ((blockIdx.x & 15) << 3) + wave;  // 0..127
  const int b    = bh >> 4;
  const int h    = bh & 15;

  const __bf16* kglob = kb + (size_t)bh * S_LEN * DH;  // [S][Dh]
  const __bf16* vglob = vt + (size_t)bh * DH * S_LEN;  // [Dh][S]

  // Stage one 32-key chunk (K: 32x64, V^T: 64x32) -> LDS buffer `buf`.
  // 256 threads x 16 B cover each 4 KiB tile exactly.
  auto stage = [&](int k0, int buf) {
    {
      const int row = tid >> 3, seg = tid & 7;  // K rows 0..31, 8 segs
      async_cp16(&kt[buf][row * DH + seg * 8],
                 kglob + (size_t)(k0 + row) * DH + seg * 8);
    }
    {
      const int row = tid >> 2, seg = tid & 3;  // V^T rows 0..63, 4 segs
      async_cp16(&vs[buf][row * 32 + seg * 8],
                 vglob + (size_t)row * S_LEN + k0 + seg * 8);
    }
  };

  // Q A-fragments: rows qblk*16..+15, Dh split into two K=32 chunks.
  const __bf16* qbase = qb + ((size_t)bh * S_LEN + qblk * 16) * DH;
  const v16bf aq0 = frag_from_bf16(qbase + lsub * DH,      khalf);
  const v16bf aq1 = frag_from_bf16(qbase + lsub * DH + 32, khalf);

  // Row m = r + 8*khalf lives in this lane; stats replicated across lsub.
  float run_max[8], run_sum[8];
  v8f o[4];
#pragma unroll
  for (int r = 0; r < 8; ++r) {
    run_max[r] = -INFINITY;
    run_sum[r] = 0.f;
#pragma unroll
    for (int d = 0; d < 4; ++d) o[d][r] = 0.f;
  }

  const float scale = 0.125f;  // 1/sqrt(Dh)

  stage(0, 0);
  async_wait_all();
  __syncthreads();

  for (int k0 = 0; k0 < S_LEN; k0 += 32) {
    const int buf = (k0 >> 5) & 1;
    if (k0 + 32 < S_LEN) stage(k0 + 32, buf ^ 1);  // overlap DMA with compute

    float st[2][8];
#pragma unroll
    for (int t = 0; t < 2; ++t) {
      const __bf16* krow = &kt[buf][(t * 16 + lsub) * DH];
      v16bf bk0 = frag_from_bf16(krow,      khalf);
      v16bf bk1 = frag_from_bf16(krow + 32, khalf);
      v8f sacc = {};
      sacc = WMMA_BF16(aq0, bk0, sacc);
      sacc = WMMA_BF16(aq1, bk1, sacc);
      // key-padding mask: this lane owns key column k0 + t*16 + lsub
      const int mk = mask[b * S_LEN + k0 + t * 16 + lsub];
#pragma unroll
      for (int r = 0; r < 8; ++r)
        st[t][r] = mk ? sacc[r] * scale : -1e10f;
    }

    // Online softmax per row; xor masks 1/2/4/8 stay within the 16-lane half.
#pragma unroll
    for (int r = 0; r < 8; ++r) {
      float cm = fmaxf(st[0][r], st[1][r]);
      cm = fmaxf(cm, __shfl_xor(cm, 1));
      cm = fmaxf(cm, __shfl_xor(cm, 2));
      cm = fmaxf(cm, __shfl_xor(cm, 4));
      cm = fmaxf(cm, __shfl_xor(cm, 8));
      const float nm   = fmaxf(run_max[r], cm);
      const float corr = __expf(run_max[r] - nm);
      run_max[r] = nm;
      const float p0 = __expf(st[0][r] - nm);
      const float p1 = __expf(st[1][r] - nm);
      st[0][r] = p0;
      st[1][r] = p1;
      float ps = p0 + p1;
      ps += __shfl_xor(ps, 1);
      ps += __shfl_xor(ps, 2);
      ps += __shfl_xor(ps, 4);
      ps += __shfl_xor(ps, 8);
      run_sum[r] = run_sum[r] * corr + ps;
#pragma unroll
      for (int d = 0; d < 4; ++d) o[d][r] *= corr;
    }

    // P: C-layout regs -> row-major LDS tile -> A-fragment (wave-local).
    __bf16* pw = plds[wave];
#pragma unroll
    for (int t = 0; t < 2; ++t)
#pragma unroll
      for (int r = 0; r < 8; ++r)
        pw[(r + 8 * khalf) * 32 + t * 16 + lsub] = (__bf16)st[t][r];
    const v16bf ap = frag_from_bf16(pw + lsub * 32, khalf);

    // O += P @ V from the staged V^T tile (contiguous along keys).
#pragma unroll
    for (int d = 0; d < 4; ++d) {
      v16bf bv = frag_from_bf16(&vs[buf][(d * 16 + lsub) * 32], khalf);
      o[d] = WMMA_BF16(ap, bv, o[d]);
    }

    if (k0 + 32 < S_LEN) async_wait_all();  // next chunk landed in buf^1
    __syncthreads();  // all waves done with buf; buf^1 visible to all
  }

  // Normalize and write bf16 to [B, S, H*Dh] (= transpose(0,2,1,3) merge).
#pragma unroll
  for (int r = 0; r < 8; ++r) {
    const float inv = 1.f / run_sum[r];
    const int s = qblk * 16 + r + 8 * khalf;
#pragma unroll
    for (int d = 0; d < 4; ++d)
      xb[((size_t)b * S_LEN + s) * 1024 + h * 64 + d * 16 + lsub] =
          (__bf16)(o[d][r] * inv);
  }
}

// ---------------------------------------------------------------------------
extern "C" void kernel_launch(void* const* d_in, const int* in_sizes, int n_in,
                              void* d_out, int out_size, void* d_ws,
                              size_t ws_size, hipStream_t stream) {
  const float* query = (const float*)d_in[0];
  const float* key_  = (const float*)d_in[1];
  const float* value = (const float*)d_in[2];
  const int*   mask  = (const int*)d_in[3];
  const float* Wq  = (const float*)d_in[4];
  const float* bq  = (const float*)d_in[5];
  const float* Wk  = (const float*)d_in[6];
  const float* bk  = (const float*)d_in[7];
  const float* Wv  = (const float*)d_in[8];
  const float* bv  = (const float*)d_in[9];
  const float* Wfc = (const float*)d_in[10];
  const float* bfc = (const float*)d_in[11];

  const int M = 4 * 2048, N = 1024, K = 1024;

  // Workspace: 4 x (B*H*S*Dh) bf16 = 64 MiB + 4 bf16 weight copies (8 MiB).
  const size_t nqk = (size_t)4 * 16 * 2048 * 64;
  const size_t nw  = (size_t)1024 * 1024;
  __bf16* qb    = (__bf16*)d_ws;
  __bf16* kbf   = qb + nqk;
  __bf16* vtb   = kbf + nqk;
  __bf16* xb    = vtb + nqk;
  __bf16* wq_b  = xb + nqk;
  __bf16* wk_b  = wq_b + nw;
  __bf16* wv_b  = wk_b + nw;
  __bf16* wfc_b = wv_b + nw;

  // One-time weight conversion to bf16 (removes per-tile cvt from GEMMs).
  const int cvtBlocks = (int)(nw / (256 * 8));
  cvt_f32_bf16<<<cvtBlocks, 256, 0, stream>>>(Wq,  wq_b,  (int)nw);
  cvt_f32_bf16<<<cvtBlocks, 256, 0, stream>>>(Wk,  wk_b,  (int)nw);
  cvt_f32_bf16<<<cvtBlocks, 256, 0, stream>>>(Wv,  wv_b,  (int)nw);
  cvt_f32_bf16<<<cvtBlocks, 256, 0, stream>>>(Wfc, wfc_b, (int)nw);

  dim3 gdim(M / 128, N / 128);  // 64 x 8 blocks of 256 threads

  gemm_wmma<false, 1><<<gdim, 256, 0, stream>>>(query, wq_b, bq, qb,  M, N, K);
  gemm_wmma<false, 1><<<gdim, 256, 0, stream>>>(key_,  wk_b, bk, kbf, M, N, K);
  gemm_wmma<false, 2><<<gdim, 256, 0, stream>>>(value, wv_b, bv, vtb, M, N, K);

  // 64 (b,h) pairs x 16 blocks each; 8 waves/block = 128 q-blocks per (b,h).
  attn_wmma<<<dim3(64 * 16), 256, 0, stream>>>(qb, kbf, vtb, mask, xb);

  gemm_wmma<true, 0><<<gdim, 256, 0, stream>>>(xb, wfc_b, bfc, d_out, M, N, K);
}
